// PEFuse_66425964200552
// MI455X (gfx1250) — compile-verified, matrix operand order checked by reading, not run on hardware
//
#include <hip/hip_runtime.h>
#include <hip/hip_bf16.h>
#include <stdint.h>

typedef _Float16 v16h __attribute__((ext_vector_type(16)));
typedef float    v8f  __attribute__((ext_vector_type(8)));
typedef float    v2f  __attribute__((ext_vector_type(2)));

#define BATCH 4
#define DIM   192
#define QKV   576
#define HWF   65536     // 256*256 full res
#define NP    16384     // 128*128 pooled
#define HEADS 6

// ---- CDNA5 async global->LDS copy (ASYNCcnt tracked) ----------------------
__device__ __forceinline__ void async_copy_b128(uint32_t lds_off, const float* gptr) {
  asm volatile("global_load_async_to_lds_b128 %0, %1, off"
               :: "v"(lds_off), "v"(gptr) : "memory");
}
__device__ __forceinline__ void wait_async0() {
  asm volatile("s_wait_asynccnt 0" ::: "memory");
}
__device__ __forceinline__ uint32_t lds_off_of(const void* p) {
  return (uint32_t)(uintptr_t)p;   // LDS aperture: low 32 bits = LDS offset
}

// ---------------------------------------------------------------------------
// K1: qkv 1x1 conv (GEMM M=576,K=192) fused with 2x2 maxpool.
// Workgroup = 9 waves, one (b, 16-pixel pooled N tile); each wave owns 4 of
// the 36 M-tiles. Per 32-wide K step: the 32ch x (2 rows x 32 cols) X slice
// (8 KB) is async-DMA'd to LDS once, then every wave builds the 4 pool-offset
// B fragments from LDS and reuses them across its 4 M tiles (16 WMMAs/step).
// X is read from HBM exactly once.
// ---------------------------------------------------------------------------
__global__ __launch_bounds__(288) void k1_qkv_pool(const float* __restrict__ x,
                                                   const float* __restrict__ w,
                                                   float* __restrict__ outp) {
  __shared__ __align__(16) float smem[2048];    // 32 k x 64 pixels

  const int t    = threadIdx.x;
  const int lane = t & 31;
  const int wv   = t >> 5;                      // 0..8
  const int nt   = blockIdx.x & 1023;
  const int b    = blockIdx.x >> 10;
  const int l16  = lane & 15;
  const int kh   = (lane >> 4) * 8;             // K-half per WMMA A/B layout
  const int py0  = nt >> 3;                     // pooled row of this N tile
  const int px0  = (nt & 7) * 16;               // first pooled col
  const long xb  = (long)b * DIM * HWF;
  const int n    = nt * 16 + l16;               // this lane's pooled pixel

  v8f acc[4][4];                                // [m-subtile][pool offset]
  #pragma unroll
  for (int mi = 0; mi < 4; ++mi)
    #pragma unroll
    for (int o = 0; o < 4; ++o) acc[mi][o] = {};

  for (int ks = 0; ks < 6; ++ks) {              // K = 192 = 6*32
    // ---- async stage: 512 x b128 transfers by first 256 threads ----
    if (t < 256) {
      #pragma unroll
      for (int e = 0; e < 2; ++e) {
        const int u   = e * 256 + t;            // 16-byte chunk id
        const int lin = u * 4;                  // float id: k*64 + pix
        const int k   = lin >> 6;
        const int pix = lin & 63;
        const int ysub = pix >> 5, xo = pix & 31;
        const float* g = x + xb + (long)(ks * 32 + k) * HWF
                           + (2 * py0 + ysub) * 256 + 2 * px0 + xo;
        async_copy_b128(lds_off_of(&smem[lin]), g);
      }
    }
    wait_async0();
    __syncthreads();

    // ---- build 4 pool-offset B fragments from LDS ----
    v16h bfr[4];
    #pragma unroll
    for (int o = 0; o < 4; ++o) {
      const int pix_l = (o >> 1) * 32 + 2 * l16 + (o & 1);
      #pragma unroll
      for (int j = 0; j < 8; ++j) {
        bfr[o][j]     = (_Float16)smem[(kh + j) * 64 + pix_l];
        bfr[o][j + 8] = (_Float16)smem[(kh + 16 + j) * 64 + pix_l];
      }
    }

    // ---- 4 M tiles per wave, B reused across all of them ----
    const int kb = ks * 32 + kh;
    #pragma unroll
    for (int mi = 0; mi < 4; ++mi) {
      const int row = (wv * 4 + mi) * 16 + l16;
      v16h a;
      #pragma unroll
      for (int j = 0; j < 8; ++j) {
        a[j]     = (_Float16)w[row * DIM + kb + j];
        a[j + 8] = (_Float16)w[row * DIM + kb + 16 + j];
      }
      #pragma unroll
      for (int o = 0; o < 4; ++o)
        acc[mi][o] = __builtin_amdgcn_wmma_f32_16x16x32_f16(
                         false, a, false, bfr[o], (short)0, acc[mi][o], false, false);
    }
    __syncthreads();                            // LDS reused next K step
  }

  #pragma unroll
  for (int mi = 0; mi < 4; ++mi)
    #pragma unroll
    for (int g = 0; g < 8; ++g) {
      const int r = (wv * 4 + mi) * 16 + g + (lane >> 4) * 8;
      float m = fmaxf(fmaxf(acc[mi][0][g], acc[mi][1][g]),
                      fmaxf(acc[mi][2][g], acc[mi][3][g]));
      outp[((long)b * QKV + r) * NP + n] = m;
    }
}

// ---------------------------------------------------------------------------
// K2: depthwise 3x3 SAME conv on [4,576,128,128]. Pure streaming.
// ---------------------------------------------------------------------------
__global__ __launch_bounds__(256) void k2_dw(const float* __restrict__ in,
                                             const float* __restrict__ w,
                                             float* __restrict__ out) {
  const long idx = (long)blockIdx.x * 256 + threadIdx.x;
  const long total = (long)BATCH * QKV * NP;
  if (idx >= total) return;
  const int n  = (int)(idx & (NP - 1));
  const long bc = idx >> 14;
  const int c  = (int)(bc % QKV);
  const int b  = (int)(bc / QKV);
  const int y  = n >> 7, x = n & 127;
  const float* base = in + ((long)b * QKV + c) * NP;
  float s = 0.f;
  #pragma unroll
  for (int i = 0; i < 3; ++i) {
    const int yy = y + i - 1;
    if (yy < 0 || yy > 127) continue;
    #pragma unroll
    for (int j = 0; j < 3; ++j) {
      const int xx = x + j - 1;
      if (xx < 0 || xx > 127) continue;
      s += w[c * 9 + i * 3 + j] * base[(yy << 7) + xx];
    }
  }
  out[idx] = s;
}

// ---------------------------------------------------------------------------
// K3: L2 row norms of q (ch 0..191) and k (ch 192..383), clamped at 1e-12.
// ---------------------------------------------------------------------------
__global__ __launch_bounds__(256) void k3_norm(const float* __restrict__ qk,
                                               float* __restrict__ norms) {
  const int rowid = blockIdx.x;              // 0..1535
  const int b = rowid / 384, ch = rowid % 384;
  const float* p = qk + ((long)b * QKV + ch) * NP;
  float s = 0.f;
  for (int i = threadIdx.x; i < NP; i += 256) { float v = p[i]; s += v * v; }
  __shared__ float red[256];
  red[threadIdx.x] = s;
  __syncthreads();
  for (int off = 128; off; off >>= 1) {
    if (threadIdx.x < off) red[threadIdx.x] += red[threadIdx.x + off];
    __syncthreads();
  }
  if (threadIdx.x == 0) norms[rowid] = fmaxf(sqrtf(red[0]), 1e-12f);
}

__global__ void k_zero(float* p, int nfl) {
  int i = blockIdx.x * 256 + threadIdx.x;
  if (i < nfl) p[i] = 0.f;
}

// ---------------------------------------------------------------------------
// K4: gram[b,h,c,d] = sum_n q[c,n]*k[d,n].  M=N=32, K=16384 per (b,h).
// fp32 WMMA 16x16x4, 16-way K-split, f32 atomics to combine partials.
// ---------------------------------------------------------------------------
__global__ __launch_bounds__(32) void k4_gram(const float* __restrict__ qk,
                                              float* __restrict__ gram) {
  const int KS = 16;
  int idb = blockIdx.x;
  const int kslice = idb % KS; idb /= KS;
  const int tile = idb % 4;
  const int bh   = idb / 4;
  const int b = bh / HEADS, h = bh % HEADS;
  const int mt = tile >> 1, ntc = tile & 1;
  const int lane = threadIdx.x;
  const int l16 = lane & 15, ko2 = (lane >> 4) * 2;
  const float* qrow = qk + ((long)b * QKV + h * 32 + mt * 16 + l16) * NP;
  const float* krow = qk + ((long)b * QKV + 192 + h * 32 + ntc * 16 + l16) * NP;
  v8f acc = {};
  const int kbeg = kslice * (NP / KS), kend = kbeg + (NP / KS);
  for (int k = kbeg; k < kend; k += 4) {
    v2f av, bv;
    av[0] = qrow[k + ko2];     av[1] = qrow[k + ko2 + 1];
    bv[0] = krow[k + ko2];     bv[1] = krow[k + ko2 + 1];
    acc = __builtin_amdgcn_wmma_f32_16x16x4_f32(
              false, av, false, bv, (short)0, acc, false, false);
  }
  float* g = gram + (long)bh * 1024;
  #pragma unroll
  for (int gi = 0; gi < 8; ++gi) {
    const int r = mt * 16 + gi + (lane >> 4) * 8;
    atomicAdd(&g[r * 32 + (ntc * 16 + l16)], acc[gi]);
  }
}

// ---------------------------------------------------------------------------
// K5: scale gram by temperature/norms, 4x top-k masked softmax, combine with
// a1..a4 into a single 32x32 matrix M per (b,h). One wave per (b,h),
// lane = row. Rank is computed branch-free with stable-topk tie semantics.
// ---------------------------------------------------------------------------
__global__ __launch_bounds__(32) void k5_attn(const float* __restrict__ gram,
                                              const float* __restrict__ norms,
                                              const float* __restrict__ temp,
                                              const float* __restrict__ a1,
                                              const float* __restrict__ a2,
                                              const float* __restrict__ a3,
                                              const float* __restrict__ a4,
                                              float* __restrict__ Mmat) {
  const int bh = blockIdx.x;
  const int b = bh / HEADS, h = bh % HEADS;
  const int c = threadIdx.x;
  const float* grow = gram + (long)bh * 1024 + c * 32;
  const float nq = norms[b * 384 + h * 32 + c];
  const float t  = temp[h];
  float r[32];
  #pragma unroll
  for (int d = 0; d < 32; ++d) {
    const float nk = norms[b * 384 + 192 + h * 32 + d];
    r[d] = grow[d] * t / (nq * nk);
  }
  int rank[32];
  #pragma unroll
  for (int d = 0; d < 32; ++d) {
    int rk = 0;
    #pragma unroll
    for (int e = 0; e < 32; ++e)
      rk += (r[e] > r[d]) || ((r[e] == r[d]) && (e < d));
    rank[d] = rk;
  }
  const int kks[4] = {16, 21, 24, 25};   // C/2, 2C/3, 3C/4, 4C/5
  const float aw[4] = {a1[0], a2[0], a3[0], a4[0]};
  float orow[32];
  #pragma unroll
  for (int d = 0; d < 32; ++d) orow[d] = 0.f;
  #pragma unroll
  for (int i = 0; i < 4; ++i) {
    const int kk = kks[i];
    float mx = -3.4e38f;
    #pragma unroll
    for (int d = 0; d < 32; ++d) if (rank[d] < kk) mx = fmaxf(mx, r[d]);
    float s = 0.f;
    #pragma unroll
    for (int d = 0; d < 32; ++d) if (rank[d] < kk) s += expf(r[d] - mx);
    const float inv = aw[i] / s;
    #pragma unroll
    for (int d = 0; d < 32; ++d) if (rank[d] < kk) orow[d] += expf(r[d] - mx) * inv;
  }
  float* Mo = Mmat + (long)bh * 1024 + c * 32;
  #pragma unroll
  for (int d = 0; d < 32; ++d) Mo[d] = orow[d];
}

// ---------------------------------------------------------------------------
// K6: g = gelu(M @ v). Block = (b,h, 64 n); thread = (c, 8 n's).
// ---------------------------------------------------------------------------
__global__ __launch_bounds__(256) void k6_av_gelu(const float* __restrict__ qk,
                                                  const float* __restrict__ Mmat,
                                                  float* __restrict__ g) {
  int idb = blockIdx.x;
  const int chunk = idb & 255;            // 16384/64 chunks
  const int bh = idb >> 8;
  const int b = bh / HEADS, h = bh % HEADS;
  __shared__ float Ml[1024];
  for (int i = threadIdx.x; i < 1024; i += 256) Ml[i] = Mmat[(long)bh * 1024 + i];
  __syncthreads();
  const int t = threadIdx.x;
  const int c = t >> 3, nsub = t & 7;
  const int n0 = chunk * 64;
  const float* v = qk + ((long)b * QKV + 384 + h * 32) * NP;
  float acc[8];
  #pragma unroll
  for (int j = 0; j < 8; ++j) acc[j] = 0.f;
  #pragma unroll
  for (int d = 0; d < 32; ++d) {
    const float m = Ml[c * 32 + d];
    #pragma unroll
    for (int j = 0; j < 8; ++j)
      acc[j] += m * v[d * NP + n0 + nsub + 8 * j];
  }
  float* go = g + ((long)b * DIM + h * 32 + c) * NP;
  #pragma unroll
  for (int j = 0; j < 8; ++j) {
    const float xv = acc[j];
    go[n0 + nsub + 8 * j] = 0.5f * xv * (1.f + erff(xv * 0.70710678118654752f));
  }
}

// ---------------------------------------------------------------------------
// K7: proj 1x1 conv (GEMM M=192,K=192) fused with nearest 2x upsample.
// Workgroup = 6 waves / one N tile; each wave owns 2 of the 12 M tiles.
// Per K step the 32x16 g slice (2 KB) is async-staged to LDS and shared.
// ---------------------------------------------------------------------------
__global__ __launch_bounds__(192) void k7_proj_up(const float* __restrict__ g,
                                                  const float* __restrict__ pw,
                                                  float* __restrict__ out) {
  __shared__ __align__(16) float smem[512];     // 32 k x 16 pixels

  const int t    = threadIdx.x;
  const int lane = t & 31;
  const int wv   = t >> 5;                      // 0..5
  const int nt   = blockIdx.x & 1023;
  const int b    = blockIdx.x >> 10;
  const int l16  = lane & 15;
  const int kh   = (lane >> 4) * 8;
  const int n0   = nt * 16;
  const int n    = n0 + l16;
  const int py = n >> 7, px = n & 127;
  const float* gb = g + (long)b * DIM * NP;

  v8f acc[2];
  acc[0] = {}; acc[1] = {};

  for (int ks = 0; ks < 6; ++ks) {
    if (t < 128) {                              // 128 x b128 = 2 KB slice
      const int k  = t >> 2;
      const int xo = (t & 3) * 4;
      async_copy_b128(lds_off_of(&smem[k * 16 + xo]),
                      gb + (long)(ks * 32 + k) * NP + n0 + xo);
    }
    wait_async0();
    __syncthreads();

    v16h bv;
    #pragma unroll
    for (int j = 0; j < 8; ++j) {
      bv[j]     = (_Float16)smem[(kh + j) * 16 + l16];
      bv[j + 8] = (_Float16)smem[(kh + 16 + j) * 16 + l16];
    }

    const int kb = ks * 32 + kh;
    #pragma unroll
    for (int mi = 0; mi < 2; ++mi) {
      const int row = (wv * 2 + mi) * 16 + l16;
      v16h a;
      #pragma unroll
      for (int j = 0; j < 8; ++j) {
        a[j]     = (_Float16)pw[row * DIM + kb + j];
        a[j + 8] = (_Float16)pw[row * DIM + kb + 16 + j];
      }
      acc[mi] = __builtin_amdgcn_wmma_f32_16x16x32_f16(
                    false, a, false, bv, (short)0, acc[mi], false, false);
    }
    __syncthreads();
  }

  #pragma unroll
  for (int mi = 0; mi < 2; ++mi)
    #pragma unroll
    for (int gi = 0; gi < 8; ++gi) {
      const int r = (wv * 2 + mi) * 16 + gi + (lane >> 4) * 8;
      const float vv = acc[mi][gi];
      const long base = (((long)b * DIM + r) * 256 + 2 * py) * 256 + 2 * px;
      out[base]       = vv;
      out[base + 1]   = vv;
      out[base + 256] = vv;
      out[base + 257] = vv;
    }
}

// ---------------------------------------------------------------------------
extern "C" void kernel_launch(void* const* d_in, const int* in_sizes, int n_in,
                              void* d_out, int out_size, void* d_ws, size_t ws_size,
                              hipStream_t stream) {
  const float* x      = (const float*)d_in[0];
  const float* temp   = (const float*)d_in[1];
  const float* qkv_w  = (const float*)d_in[2];
  const float* dw_w   = (const float*)d_in[3];
  const float* proj_w = (const float*)d_in[4];
  const float* a1 = (const float*)d_in[5];
  const float* a2 = (const float*)d_in[6];
  const float* a3 = (const float*)d_in[7];
  const float* a4 = (const float*)d_in[8];
  float* out = (float*)d_out;

  float* ws    = (float*)d_ws;
  const long TQKV = (long)BATCH * QKV * NP;       // 37,748,736 floats
  float* qkvp  = ws;                               // pooled qkv (f32)
  float* qkvdw = ws + TQKV;                        // after depthwise
  float* norms = qkvdw + TQKV;                     // 1536 floats
  float* gram  = norms + 1536;                     // 24576 floats
  float* Mm    = gram + 24576;                     // 24576 floats
  float* gbuf  = qkvp;                             // reuse region 0 for gelu(Mv)

  // 1) qkv conv + maxpool: one block per (b, pooled 16-pixel tile)
  k1_qkv_pool<<<4096, 288, 0, stream>>>(x, qkv_w, qkvp);
  // 2) depthwise 3x3
  k2_dw<<<(int)(TQKV / 256), 256, 0, stream>>>(qkvp, dw_w, qkvdw);
  // 3) q/k norms
  k3_norm<<<1536, 256, 0, stream>>>(qkvdw, norms);
  // 4) gram (zero first: ws is poisoned by harness)
  k_zero<<<(24576 + 255) / 256, 256, 0, stream>>>(gram, 24576);
  k4_gram<<<24 * 4 * 16, 32, 0, stream>>>(qkvdw, gram);
  // 5) combined masked-softmax matrix
  k5_attn<<<24, 32, 0, stream>>>(gram, norms, temp, a1, a2, a3, a4, Mm);
  // 6) M @ v + gelu
  k6_av_gelu<<<24 * 256, 256, 0, stream>>>(qkvdw, Mm, gbuf);
  // 7) upsample + proj conv
  k7_proj_up<<<4096, 192, 0, stream>>>(gbuf, proj_w, out);
}
